// GraphConvolution_24592982737579
// MI455X (gfx1250) — matrix operand order repaired and checked by reading, not compiled, not run on hardware
//
#include <hip/hip_runtime.h>

typedef __attribute__((ext_vector_type(16))) __bf16 v16bf;
typedef __attribute__((ext_vector_type(8)))  __bf16 v8bf;
typedef __attribute__((ext_vector_type(4)))  __bf16 v4bf;
typedef __attribute__((ext_vector_type(8)))  float  v8f;

#define IN_DIM  512
#define OUT_DIM 256
#define KTILES  (IN_DIM / 32)    // 16 k-tiles of 32
#define NTILES  (OUT_DIM / 16)   // 16 n-tiles of 16
#define MTILE   64               // rows per workgroup (4 WMMA subtiles)
#define MSUB    (MTILE / 16)
#define WFRAG_ELEMS (IN_DIM * OUT_DIM)   // 131072 bf16 per hi/lo plane

// ---------------------------------------------------------------------------
// Pack W (fp32 [512,256]) into bf16 hi/lo B-fragment layout.
// Fragment layout (this code's contiguous k order, B 32x16): lane n(0..15) holds col n,
// K=kt*32+0..15 ; lane 16+n holds col n, K=kt*32+16..31.  Linearized so a
// wave reading tile (kt,nt) loads 16 contiguous bf16 per lane.
// ---------------------------------------------------------------------------
__global__ void pack_w_kernel(const float* __restrict__ w,
                              __bf16* __restrict__ w_hi,
                              __bf16* __restrict__ w_lo) {
  int t  = blockIdx.x * blockDim.x + threadIdx.x;   // 0 .. 131071
  int j  = t & 15;
  int l  = (t >> 4) & 31;
  int nt = (t >> 9) & 15;
  int kt = t >> 13;
  int k = kt * 32 + (l >> 4) * 16 + j;
  int n = nt * 16 + (l & 15);
  float v = w[k * OUT_DIM + n];
  __bf16 h = (__bf16)v;
  w_hi[t] = h;
  w_lo[t] = (__bf16)(v - (float)h);
}

__global__ void zero_int_kernel(int* __restrict__ p, int total) {
  int i = blockIdx.x * blockDim.x + threadIdx.x;
  if (i < total) p[i] = 0;
}

// Histogram of destination rows (int atomics -> global_atomic_add_u32).
__global__ void count_kernel(const int* __restrict__ erow,
                             int* __restrict__ cnt, int n_edges) {
  int e = blockIdx.x * blockDim.x + threadIdx.x;
  if (e < n_edges) atomicAdd(&cnt[erow[e]], 1);
}

// ---------------------------------------------------------------------------
// Single-workgroup exclusive scan over n counters (n ~ 100k).
// Per 1024-chunk: wave32 __shfl_up scans (no barriers) + 32-entry LDS combine.
// ---------------------------------------------------------------------------
__global__ __launch_bounds__(1024)
void scan_kernel(const int* __restrict__ cnt, int* __restrict__ row_start, int n) {
  __shared__ int wsum[32];
  __shared__ int carry_s;
  const int tid  = threadIdx.x;
  const int lane = tid & 31;
  const int w    = tid >> 5;
  if (tid == 0) carry_s = 0;
  __syncthreads();
  for (int base = 0; base < n; base += 1024) {
    int i = base + tid;
    int v = (i < n) ? cnt[i] : 0;
    int s = v;
#pragma unroll
    for (int off = 1; off < 32; off <<= 1) {
      int t = __shfl_up(s, off, 32);
      if (lane >= off) s += t;
    }
    if (lane == 31) wsum[w] = s;
    __syncthreads();
    if (w == 0) {
      int ws = wsum[lane];
#pragma unroll
      for (int off = 1; off < 32; off <<= 1) {
        int t = __shfl_up(ws, off, 32);
        if (lane >= off) ws += t;
      }
      wsum[lane] = ws;                 // inclusive scan of wave sums
    }
    __syncthreads();
    int woff = (w == 0) ? 0 : wsum[w - 1];
    int c = carry_s;
    if (i < n) row_start[i] = c + woff + s - v;   // exclusive
    int tot = wsum[31];
    __syncthreads();
    if (tid == 0) carry_s = c + tot;
    __syncthreads();
  }
  if (tid == 0) row_start[n] = carry_s;
}

// Scatter edges into CSR slots (atomic cursor per row).
__global__ void fill_kernel(const int* __restrict__ erow,
                            const int* __restrict__ ecol,
                            const float* __restrict__ ev,
                            int* __restrict__ cursor,
                            int* __restrict__ csr_col,
                            float* __restrict__ csr_val, int n_edges) {
  int e = blockIdx.x * blockDim.x + threadIdx.x;
  if (e < n_edges) {
    int pos = atomicAdd(&cursor[erow[e]], 1);
    csr_col[pos] = ecol[e];
    csr_val[pos] = ev[e];
  }
}

// ---------------------------------------------------------------------------
// Dense GEMM: xw = x @ W, bf16 hi/lo split via v_wmma_f32_16x16x32_bf16.
// One workgroup = 512 threads = 16 waves; computes 64 rows x 256 cols.
// The 64x512 A slab lives in LDS (bf16 hi+lo, 128 KB); each wave owns one
// 16-col N-tile and reuses its B fragments across 4 M-subtiles, cutting
// B-fragment cache traffic 4x vs a 16-row block.
// ---------------------------------------------------------------------------
__global__ __launch_bounds__(512)
void gemm_kernel(const float* __restrict__ x,
                 const __bf16* __restrict__ w_hi,
                 const __bf16* __restrict__ w_lo,
                 float* __restrict__ xw, int n_nodes) {
  __shared__ __bf16 lds_ah[MTILE * IN_DIM];   // 64 KB
  __shared__ __bf16 lds_al[MTILE * IN_DIM];   // 64 KB

  const int R0  = blockIdx.x * MTILE;
  const int tid = threadIdx.x;

  // Stage + split: float4 x reads, v4bf (b64) LDS stores.
  const float4* x4 = (const float4*)x;
  for (int i = tid; i < MTILE * IN_DIM / 4; i += 512) {
    int rr  = i >> 7;                 // row within slab (128 float4 per row)
    int kf  = i & 127;
    int row = R0 + rr;
    if (row >= n_nodes) row = n_nodes - 1;   // clamp (uniform EXEC)
    float4 v = x4[(size_t)row * (IN_DIM / 4) + kf];
    v4bf h, l;
    h[0] = (__bf16)v.x; h[1] = (__bf16)v.y; h[2] = (__bf16)v.z; h[3] = (__bf16)v.w;
    l[0] = (__bf16)(v.x - (float)h[0]);
    l[1] = (__bf16)(v.y - (float)h[1]);
    l[2] = (__bf16)(v.z - (float)h[2]);
    l[3] = (__bf16)(v.w - (float)h[3]);
    *(v4bf*)&lds_ah[rr * IN_DIM + kf * 4] = h;
    *(v4bf*)&lds_al[rr * IN_DIM + kf * 4] = l;
  }
  __syncthreads();

  const int nt   = tid >> 5;          // wave id == N-tile
  const int lane = tid & 31;
  // A fragment addressing (contiguous k order, same as packed B):
  //   lane L(0..15): row L, halfs K {0..15}
  //   lane 16+L    : row L, halfs K {16..31}
  const int arow = lane & 15;
  const int koff = (lane >> 4) * 16;

  v8f acc[MSUB] = {};
  for (int kt = 0; kt < KTILES; ++kt) {
    const int bidx = ((kt * NTILES + nt) * 32 + lane) * 16;
    v16bf b_hi = *(const v16bf*)&w_hi[bidx];
    v16bf b_lo = *(const v16bf*)&w_lo[bidx];
#pragma unroll
    for (int m = 0; m < MSUB; ++m) {
      const int abase = (m * 16 + arow) * IN_DIM + kt * 32 + koff;
      v8bf ah0 = *(const v8bf*)&lds_ah[abase];
      v8bf ah1 = *(const v8bf*)&lds_ah[abase + 8];
      v8bf al0 = *(const v8bf*)&lds_al[abase];
      v8bf al1 = *(const v8bf*)&lds_al[abase + 8];
      v16bf a_hi = __builtin_shufflevector(ah0, ah1, 0,1,2,3,4,5,6,7,8,9,10,11,12,13,14,15);
      v16bf a_lo = __builtin_shufflevector(al0, al1, 0,1,2,3,4,5,6,7,8,9,10,11,12,13,14,15);

      acc[m] = __builtin_amdgcn_wmma_f32_16x16x32_bf16(false, a_hi, false, b_hi,
                                                       (short)0, acc[m], false, false);
      acc[m] = __builtin_amdgcn_wmma_f32_16x16x32_bf16(false, a_hi, false, b_lo,
                                                       (short)0, acc[m], false, false);
      acc[m] = __builtin_amdgcn_wmma_f32_16x16x32_bf16(false, a_lo, false, b_hi,
                                                       (short)0, acc[m], false, false);
    }
  }

  // C/D layout: VGPR i -> row i + 8*(lane>=16); col = lane&15.
  const int nn = nt * 16 + (lane & 15);
#pragma unroll
  for (int m = 0; m < MSUB; ++m) {
    const int rbase = R0 + m * 16 + 8 * (lane >> 4);
    float* o = xw + (size_t)rbase * OUT_DIM + nn;
#pragma unroll
    for (int i = 0; i < 8; ++i) {
      if (rbase + i < n_nodes) o[(size_t)i * OUT_DIM] = acc[m][i];
    }
  }
}

// ---------------------------------------------------------------------------
// CSR gather SpMM + bias + ReLU: one wave per output row.  xw (102 MB) is
// L2-resident; per edge the wave streams one 1 KB row from L2, accumulates in
// registers, stores the output row exactly once.  No f32 atomics anywhere.
// ---------------------------------------------------------------------------
__global__ __launch_bounds__(256)
void spmm_csr_kernel(const int* __restrict__ row_start,
                     const int* __restrict__ csr_col,
                     const float* __restrict__ csr_val,
                     const float* __restrict__ xw,
                     const float* __restrict__ b,
                     float* __restrict__ out, int n_nodes) {
  const int lane = threadIdx.x & 31;
  const int row  = blockIdx.x * 8 + (threadIdx.x >> 5);
  if (row >= n_nodes) return;
  const int s = row_start[row];
  const int e = row_start[row + 1];

  float4 a0 = {0.f, 0.f, 0.f, 0.f};
  float4 a1 = {0.f, 0.f, 0.f, 0.f};
  for (int j = s; j < e; ++j) {
    if (j + 1 < e) {   // prefetch next gathered row (global_prefetch_b8)
      __builtin_prefetch(xw + (size_t)csr_col[j + 1] * OUT_DIM + lane * 8, 0, 0);
    }
    const int   c = csr_col[j];
    const float v = csr_val[j];
    const float4* p = (const float4*)(xw + (size_t)c * OUT_DIM);
    float4 f0 = p[lane];
    float4 f1 = p[32 + lane];
    a0.x += v * f0.x; a0.y += v * f0.y; a0.z += v * f0.z; a0.w += v * f0.w;
    a1.x += v * f1.x; a1.y += v * f1.y; a1.z += v * f1.z; a1.w += v * f1.w;
  }

  const float4* b4 = (const float4*)b;
  float4 b0 = b4[lane];
  float4 b1 = b4[32 + lane];
  a0.x = fmaxf(a0.x + b0.x, 0.f); a0.y = fmaxf(a0.y + b0.y, 0.f);
  a0.z = fmaxf(a0.z + b0.z, 0.f); a0.w = fmaxf(a0.w + b0.w, 0.f);
  a1.x = fmaxf(a1.x + b1.x, 0.f); a1.y = fmaxf(a1.y + b1.y, 0.f);
  a1.z = fmaxf(a1.z + b1.z, 0.f); a1.w = fmaxf(a1.w + b1.w, 0.f);

  float4* o4 = (float4*)(out + (size_t)row * OUT_DIM);
  o4[lane]      = a0;
  o4[32 + lane] = a1;
}

// ---------------------------------------------------------------------------
extern "C" void kernel_launch(void* const* d_in, const int* in_sizes, int n_in,
                              void* d_out, int out_size, void* d_ws, size_t ws_size,
                              hipStream_t stream) {
  const float* x    = (const float*)d_in[0];
  const int*   erow = (const int*)d_in[1];
  const int*   ecol = (const int*)d_in[2];
  const float* ev   = (const float*)d_in[3];
  const float* w    = (const float*)d_in[4];
  const float* b    = (const float*)d_in[5];

  const int n_nodes = in_sizes[0] / IN_DIM;
  const int n_edges = in_sizes[1];

  // ---- workspace layout (256B aligned blocks) ----
  char* base = (char*)d_ws;
  size_t off = 0;
  auto take = [&](size_t bytes) -> char* {
    char* p = base + off;
    off = (off + bytes + 255) & ~(size_t)255;
    return p;
  };
  __bf16* w_hi      = (__bf16*)take((size_t)WFRAG_ELEMS * 2);
  __bf16* w_lo      = (__bf16*)take((size_t)WFRAG_ELEMS * 2);
  float*  xw        = (float*) take((size_t)n_nodes * OUT_DIM * 4);
  int*    cnt       = (int*)   take((size_t)n_nodes * 4);          // reused as cursor
  int*    row_start = (int*)   take((size_t)(n_nodes + 1) * 4);
  int*    csr_col   = (int*)   take((size_t)n_edges * 4);
  float*  csr_val   = (float*) take((size_t)n_edges * 4);

  // ---- CSR build ----
  zero_int_kernel<<<(n_nodes + 255) / 256, 256, 0, stream>>>(cnt, n_nodes);
  count_kernel<<<(n_edges + 255) / 256, 256, 0, stream>>>(erow, cnt, n_edges);
  scan_kernel<<<1, 1024, 0, stream>>>(cnt, row_start, n_nodes);
  hipMemcpyAsync(cnt, row_start, (size_t)n_nodes * 4,
                 hipMemcpyDeviceToDevice, stream);                  // cursor = row_start
  fill_kernel<<<(n_edges + 255) / 256, 256, 0, stream>>>(erow, ecol, ev, cnt,
                                                         csr_col, csr_val, n_edges);

  // ---- dense GEMM (WMMA) ----
  pack_w_kernel<<<WFRAG_ELEMS / 256, 256, 0, stream>>>(w, w_hi, w_lo);
  gemm_kernel<<<(n_nodes + MTILE - 1) / MTILE, 512, 0, stream>>>(x, w_hi, w_lo, xw, n_nodes);

  // ---- CSR gather + bias + ReLU ----
  spmm_csr_kernel<<<(n_nodes + 7) / 8, 256, 0, stream>>>(row_start, csr_col, csr_val,
                                                         xw, b, (float*)d_out, n_nodes);
}
